// HeteroGNNFraudDetector_78829829750888
// MI455X (gfx1250) — matrix operand order repaired and checked by reading, not compiled, not run on hardware
//
#include <hip/hip_runtime.h>
#include <math.h>

typedef __attribute__((ext_vector_type(16))) _Float16 v16h;
typedef __attribute__((ext_vector_type(8)))  _Float16 v8h;
typedef __attribute__((ext_vector_type(8)))  float    v8f;

#define HDIM 128
#define EPS  1e-5f

// ---------------------------------------------------------------- utilities

__global__ void zero_f32_kernel(float* __restrict__ p, int n) {
    int i = blockIdx.x * blockDim.x + threadIdx.x;
    if (i < n) p[i] = 0.0f;
}

__global__ void count_kernel(const int* __restrict__ dst, float* __restrict__ cnt, int E) {
    int e = blockIdx.x * blockDim.x + threadIdx.x;
    if (e < E) atomicAdd(&cnt[dst[e]], 1.0f);
}

// scatter-sum of h_src rows onto agg rows: one thread = (edge, 4-col chunk)
__global__ void scatter_add_kernel(const float* __restrict__ hsrc,
                                   const int* __restrict__ src,
                                   const int* __restrict__ dst,
                                   float* __restrict__ agg, int E) {
    int tid = blockIdx.x * blockDim.x + threadIdx.x;
    int e = tid >> 5;            // 32 chunks of 4 cols = 128
    int c = (tid & 31) << 2;
    if (e >= E) return;
    int s = src[e];
    int d = dst[e];
    float4 v = *reinterpret_cast<const float4*>(&hsrc[(size_t)s * HDIM + c]);
    float* base = &agg[(size_t)d * HDIM + c];
    atomicAdd(base + 0, v.x);
    atomicAdd(base + 1, v.y);
    atomicAdd(base + 2, v.z);
    atomicAdd(base + 3, v.w);
}

// Pack W[K x N] f32 row-major (zero rows beyond Ksrc) into B-fragment-major f16:
//   dst[((kc*(N/16)+nt)*32 + lane)*16 + j] = W[(kc*32 + 16*(lane>=16) + j)*N + nt*16 + (lane&15)]
// so a wave's B fragment for (kc, nt) is one contiguous 32B chunk per lane.
__global__ void pack_b_kernel(const float* __restrict__ W, _Float16* __restrict__ dst,
                              int Ksrc, int Kpad, int N) {
    int i = blockIdx.x * blockDim.x + threadIdx.x;
    int total = Kpad * N;
    if (i >= total) return;
    int j    = i & 15;
    int lane = (i >> 4) & 31;
    int tile = i >> 9;                 // kc*(N/16) + nt
    int ntiles = N >> 4;
    int kc = tile / ntiles;
    int nt = tile - kc * ntiles;
    int k   = kc * 32 + ((lane >> 4) << 4) + j;
    int col = nt * 16 + (lane & 15);
    float v = (k < Ksrc) ? W[k * N + col] : 0.0f;
    dst[i] = (_Float16)v;
}

// input projection: h[N,128] = x[N,32] @ W[32,128] + b   (one block per row)
__global__ __launch_bounds__(128) void proj_kernel(const float* __restrict__ x,
                                                   const float* __restrict__ W,
                                                   const float* __restrict__ b,
                                                   float* __restrict__ h, int N) {
    __shared__ float sx[32];
    int row = blockIdx.x;
    if (row >= N) return;
    int col = threadIdx.x;
    if (col < 32) sx[col] = x[(size_t)row * 32 + col];
    __syncthreads();
    float acc = b[col];
#pragma unroll
    for (int k = 0; k < 32; ++k) acc += sx[k] * W[k * HDIM + col];
    h[(size_t)row * HDIM + col] = acc;
}

// A-fragment for v_wmma_f32_16x16x32_f16 from an f16 LDS tile (row-major, ld halves):
// lane's 16 halves are K = kb+16*hi+{0..15} -> two b128 DS loads.
__device__ __forceinline__ v16h frag_a(const _Float16* base, int ld, int row, int kb, int hi) {
    v8h lo  = *reinterpret_cast<const v8h*>(base + row * ld + kb + (hi << 4));
    v8h hi8 = *reinterpret_cast<const v8h*>(base + row * ld + kb + 8 + (hi << 4));
    return __builtin_shufflevector(lo, hi8, 0, 1, 2, 3, 4, 5, 6, 7,
                                   8, 9, 10, 11, 12, 13, 14, 15);
}

// ------------------------------------------------- SAGE layer update (WMMA)
// h[row] = relu(bn(agg_norm @ Wl + bl + h_old @ Wr)) + h_old   (in place)
// block = 128 threads = 4 waves; block owns 16 rows; wave owns 2 N-tiles.
__global__ __launch_bounds__(128) void sage_update_kernel(
    const float* __restrict__ agg, const float* __restrict__ cnt,
    float* __restrict__ h,
    const _Float16* __restrict__ Wl, const float* __restrict__ bl,
    const _Float16* __restrict__ Wr,
    const float* __restrict__ bng, const float* __restrict__ bnb,
    const float* __restrict__ bnm, const float* __restrict__ bnv,
    int N)
{
    __shared__ __align__(16) _Float16 sA[16 * HDIM];  // mean-normalized agg rows (f16)
    __shared__ __align__(16) _Float16 sH[16 * HDIM];  // old h rows (f16, GEMM operand)
    __shared__ float sHf[16 * HDIM];                  // old h rows (f32, residual)
    int rowbase = blockIdx.x * 16;
    int tid = threadIdx.x;

    for (int idx = tid; idx < 16 * HDIM; idx += 128) {
        int r = idx >> 7, c = idx & 127;
        int row = rowbase + r;
        float av = 0.0f, hv = 0.0f;
        if (row < N) {
            float inv = 1.0f / fmaxf(cnt[row], 1.0f);
            av = agg[(size_t)row * HDIM + c] * inv;
            hv = h[(size_t)row * HDIM + c];
        }
        sA[idx]  = (_Float16)av;
        sH[idx]  = (_Float16)hv;
        sHf[idx] = hv;
    }
    __syncthreads();

    int wave = tid >> 5;
    int lane = tid & 31;
    int lrow = lane & 15;   // M for A-frag, N for B/C-frags
    int hi   = lane >> 4;   // lane-group select

    const v16h* Wlv = reinterpret_cast<const v16h*>(Wl);
    const v16h* Wrv = reinterpret_cast<const v16h*>(Wr);

    for (int t = 0; t < 2; ++t) {
        int nt  = wave + t * 4;            // N-tile 0..7
        int col = nt * 16 + lrow;
        v8f acc = {};
#pragma unroll
        for (int kc = 0; kc < 4; ++kc) {
            int kb = kc * 32;
            v16h a = frag_a(sA, HDIM, lrow, kb, hi);
            v16h b = Wlv[((kc * 8 + nt) << 5) + lane];
            acc = __builtin_amdgcn_wmma_f32_16x16x32_f16(false, a, false, b,
                                                         (short)0, acc, false, false);
            a = frag_a(sH, HDIM, lrow, kb, hi);
            b = Wrv[((kc * 8 + nt) << 5) + lane];
            acc = __builtin_amdgcn_wmma_f32_16x16x32_f16(false, a, false, b,
                                                         (short)0, acc, false, false);
        }
        // epilogue: + bl, BN(eval), ReLU, + residual
        float g = bng[col], bb = bnb[col], m = bnm[col];
        float rstd = rsqrtf(bnv[col] + EPS);
        float bias = bl[col];
#pragma unroll
        for (int r8 = 0; r8 < 8; ++r8) {
            int r = r8 + hi * 8;           // C layout: M = r8 + 8*hi
            int row = rowbase + r;
            float v = acc[r8] + bias;
            v = (v - m) * rstd * g + bb;
            v = fmaxf(v, 0.0f) + sHf[r * HDIM + col];
            if (row < N) h[(size_t)row * HDIM + col] = v;
        }
    }
}

// ------------------------------------------------- edge predictor (WMMA)
// block = 128 threads = 4 waves; 16 edges per block.
__global__ __launch_bounds__(128) void edge_pred_kernel(
    const float* __restrict__ h_u, const float* __restrict__ h_r,
    const float* __restrict__ eattr,
    const int* __restrict__ src_idx, const int* __restrict__ dst_idx,
    const _Float16* __restrict__ W1, const float* __restrict__ b1,
    const float* __restrict__ bng, const float* __restrict__ bnb,
    const float* __restrict__ bnm, const float* __restrict__ bnv,
    const _Float16* __restrict__ W2, const float* __restrict__ b2,
    const float* __restrict__ W3, const float* __restrict__ b3,
    float* __restrict__ out, int E)
{
    __shared__ __align__(16) _Float16 sX[16 * 288];   // [src | dst | eattr | pad] f16
    __shared__ __align__(16) _Float16 sH1[16 * 128];
    __shared__ float sH2[16 * 64];
    int ebase = blockIdx.x * 16;
    int tid = threadIdx.x;

    for (int idx = tid; idx < 16 * 288; idx += 128) {
        int r = idx / 288, c = idx % 288;
        int e = ebase + r;
        float v = 0.0f;
        if (e < E) {
            if (c < 128)       v = h_u[(size_t)src_idx[e] * HDIM + c];
            else if (c < 256)  v = h_r[(size_t)dst_idx[e] * HDIM + (c - 128)];
            else if (c < 272)  v = eattr[(size_t)e * 16 + (c - 256)];
        }
        sX[idx] = (_Float16)v;
    }
    __syncthreads();

    int wave = tid >> 5;
    int lane = tid & 31;
    int lrow = lane & 15;
    int hi   = lane >> 4;

    const v16h* W1v = reinterpret_cast<const v16h*>(W1);
    const v16h* W2v = reinterpret_cast<const v16h*>(W2);

    // GEMM1: [16,288] @ [288,128], then ReLU then BN(eval), store f16
    for (int t = 0; t < 2; ++t) {
        int nt  = wave + t * 4;
        int col = nt * 16 + lrow;
        v8f acc = {};
#pragma unroll
        for (int kc = 0; kc < 9; ++kc) {
            v16h a = frag_a(sX, 288, lrow, kc * 32, hi);
            v16h b = W1v[((kc * 8 + nt) << 5) + lane];
            acc = __builtin_amdgcn_wmma_f32_16x16x32_f16(false, a, false, b,
                                                         (short)0, acc, false, false);
        }
        float g = bng[col], bb = bnb[col], m = bnm[col];
        float rstd = rsqrtf(bnv[col] + EPS);
        float bias = b1[col];
#pragma unroll
        for (int r8 = 0; r8 < 8; ++r8) {
            int r = r8 + hi * 8;
            float v = fmaxf(acc[r8] + bias, 0.0f);   // ReLU first (reference order)
            v = (v - m) * rstd * g + bb;             // then BN(eval)
            sH1[r * HDIM + col] = (_Float16)v;
        }
    }
    __syncthreads();

    // GEMM2: [16,128] @ [128,64] + b2, ReLU; wave w owns N-tile w (64 cols total)
    {
        int col = wave * 16 + lrow;
        v8f acc = {};
#pragma unroll
        for (int kc = 0; kc < 4; ++kc) {
            v16h a = frag_a(sH1, HDIM, lrow, kc * 32, hi);
            v16h b = W2v[((kc * 4 + wave) << 5) + lane];
            acc = __builtin_amdgcn_wmma_f32_16x16x32_f16(false, a, false, b,
                                                         (short)0, acc, false, false);
        }
        float bias = b2[col];
#pragma unroll
        for (int r8 = 0; r8 < 8; ++r8) {
            int r = r8 + hi * 8;
            sH2[r * 64 + col] = fmaxf(acc[r8] + bias, 0.0f);
        }
    }
    __syncthreads();

    // final 64-dot + sigmoid per edge
    if (tid < 16) {
        int e = ebase + tid;
        if (e < E) {
            float s = b3[0];
#pragma unroll
            for (int k = 0; k < 64; ++k) s += sH2[tid * 64 + k] * W3[k];
            out[e] = 1.0f / (1.0f + expf(-s));
        }
    }
}

// ---------------------------------------------------------------- launcher

static inline int cdiv(long long a, long long b) { return (int)((a + b - 1) / b); }

extern "C" void kernel_launch(void* const* d_in, const int* in_sizes, int n_in,
                              void* d_out, int out_size, void* d_ws, size_t ws_size,
                              hipStream_t stream) {
    const int Nu = in_sizes[0] / 32;
    const int Nr = in_sizes[1] / 32;
    const int E  = in_sizes[2] / 16;

    const float* x_user = (const float*)d_in[0];
    const float* x_rec  = (const float*)d_in[1];
    const float* eattr  = (const float*)d_in[2];
    const int*   ut     = (const int*)d_in[3];
    const int*   ru     = (const int*)d_in[4];
    const int* ut_src = ut;      const int* ut_dst = ut + E;
    const int* ru_src = ru;      const int* ru_dst = ru + E;
    const float* proj_uW = (const float*)d_in[5];
    const float* proj_ub = (const float*)d_in[6];
    const float* proj_rW = (const float*)d_in[7];
    const float* proj_rb = (const float*)d_in[8];
    const float* sage_Wl = (const float*)d_in[9];
    const float* sage_bl = (const float*)d_in[10];
    const float* sage_Wr = (const float*)d_in[11];
    const float* bn_g = (const float*)d_in[12];
    const float* bn_b = (const float*)d_in[13];
    const float* bn_m = (const float*)d_in[14];
    const float* bn_v = (const float*)d_in[15];
    const float* ep_W1 = (const float*)d_in[16];
    const float* ep_b1 = (const float*)d_in[17];
    const float* ep_bng = (const float*)d_in[18];
    const float* ep_bnb = (const float*)d_in[19];
    const float* ep_bnm = (const float*)d_in[20];
    const float* ep_bnv = (const float*)d_in[21];
    const float* ep_W2 = (const float*)d_in[22];
    const float* ep_b2 = (const float*)d_in[23];
    const float* ep_W3 = (const float*)d_in[24];
    const float* ep_b3 = (const float*)d_in[25];
    float* out = (float*)d_out;

    // workspace partition (256B aligned)
    size_t off = 0;
    auto alloc = [&](size_t bytes) -> char* {
        char* p = (char*)d_ws + off;
        off = (off + bytes + 255) & ~(size_t)255;
        return p;
    };
    float*    h_u   = (float*)alloc((size_t)Nu * HDIM * 4);
    float*    h_r   = (float*)alloc((size_t)Nr * HDIM * 4);
    float*    agg_r = (float*)alloc((size_t)Nr * HDIM * 4);
    float*    agg_u = (float*)alloc((size_t)Nu * HDIM * 4);
    float*    cnt_r = (float*)alloc((size_t)Nr * 4);
    float*    cnt_u = (float*)alloc((size_t)Nu * 4);
    _Float16* wl16  = (_Float16*)alloc((size_t)2 * 2 * HDIM * HDIM * 2);  // packed
    _Float16* wr16  = (_Float16*)alloc((size_t)2 * 2 * HDIM * HDIM * 2);  // packed
    _Float16* w1_16 = (_Float16*)alloc((size_t)288 * HDIM * 2);           // packed, K pad
    _Float16* w2_16 = (_Float16*)alloc((size_t)HDIM * 64 * 2);            // packed
    (void)ws_size;

    const int T = 256;

    // degrees (layer invariant)
    zero_f32_kernel<<<cdiv(Nr, T), T, 0, stream>>>(cnt_r, Nr);
    zero_f32_kernel<<<cdiv(Nu, T), T, 0, stream>>>(cnt_u, Nu);
    count_kernel<<<cdiv(E, T), T, 0, stream>>>(ut_dst, cnt_r, E);
    count_kernel<<<cdiv(E, T), T, 0, stream>>>(ru_dst, cnt_u, E);

    // pack weights into B-fragment-major f16
    for (int m = 0; m < 4; ++m) {
        size_t o = (size_t)m * HDIM * HDIM;
        pack_b_kernel<<<cdiv(HDIM * HDIM, T), T, 0, stream>>>(sage_Wl + o, wl16 + o, HDIM, HDIM, HDIM);
        pack_b_kernel<<<cdiv(HDIM * HDIM, T), T, 0, stream>>>(sage_Wr + o, wr16 + o, HDIM, HDIM, HDIM);
    }
    pack_b_kernel<<<cdiv(288 * HDIM, T), T, 0, stream>>>(ep_W1, w1_16, 272, 288, HDIM);
    pack_b_kernel<<<cdiv(HDIM * 64, T), T, 0, stream>>>(ep_W2, w2_16, HDIM, HDIM, 64);

    // input projections
    proj_kernel<<<Nu, 128, 0, stream>>>(x_user, proj_uW, proj_ub, h_u, Nu);
    proj_kernel<<<Nr, 128, 0, stream>>>(x_rec,  proj_rW, proj_rb, h_r, Nr);

    for (int layer = 0; layer < 2; ++layer) {
        // aggregate BOTH directions from the pre-update features
        zero_f32_kernel<<<cdiv((long long)Nr * HDIM, T), T, 0, stream>>>(agg_r, Nr * HDIM);
        zero_f32_kernel<<<cdiv((long long)Nu * HDIM, T), T, 0, stream>>>(agg_u, Nu * HDIM);
        scatter_add_kernel<<<cdiv((long long)E * 32, T), T, 0, stream>>>(h_u, ut_src, ut_dst, agg_r, E);
        scatter_add_kernel<<<cdiv((long long)E * 32, T), T, 0, stream>>>(h_r, ru_src, ru_dst, agg_u, E);

        size_t wofs0 = (size_t)(layer * 2 + 0) * HDIM * HDIM;  // edge type 0 -> recipients
        size_t wofs1 = (size_t)(layer * 2 + 1) * HDIM * HDIM;  // edge type 1 -> users
        size_t b0  = (size_t)(layer * 2 + 0) * HDIM;
        size_t b1o = (size_t)(layer * 2 + 1) * HDIM;
        size_t bnU = (size_t)(layer * 2 + 0) * HDIM;  // node type 0 = user
        size_t bnR = (size_t)(layer * 2 + 1) * HDIM;  // node type 1 = recipient

        sage_update_kernel<<<cdiv(Nr, 16), 128, 0, stream>>>(
            agg_r, cnt_r, h_r, wl16 + wofs0, sage_bl + b0, wr16 + wofs0,
            bn_g + bnR, bn_b + bnR, bn_m + bnR, bn_v + bnR, Nr);
        sage_update_kernel<<<cdiv(Nu, 16), 128, 0, stream>>>(
            agg_u, cnt_u, h_u, wl16 + wofs1, sage_bl + b1o, wr16 + wofs1,
            bn_g + bnU, bn_b + bnU, bn_m + bnU, bn_v + bnU, Nu);
    }

    edge_pred_kernel<<<cdiv(E, 16), 128, 0, stream>>>(
        h_u, h_r, eattr, ut_src, ut_dst,
        w1_16, ep_b1, ep_bng, ep_bnb, ep_bnm, ep_bnv,
        w2_16, ep_b2, ep_W3, ep_b3, out, E);
}
